// GLAM_79130477461751
// MI455X (gfx1250) — matrix operand order repaired and checked by reading, not compiled, not run on hardware
//
#include <hip/hip_runtime.h>
#include <hip/hip_bf16.h>

#define DI __device__ __forceinline__

typedef __attribute__((ext_vector_type(16))) _Float16 v16h;
typedef __attribute__((ext_vector_type(8)))  _Float16 h8;
typedef __attribute__((ext_vector_type(8)))  float    v8f;

constexpr int Bn   = 16;
constexpr int Cc   = 256;
constexpr int Hh   = 32;
constexpr int Ww   = 32;
constexpr int HW   = 1024;
constexpr int Ee   = 64;
constexpr int CO   = 256;
constexpr int CHW  = Cc * HW;    // 262144
constexpr int EHW  = Ee * HW;    // 65536
constexpr int CC   = Cc * Cc;    // 65536
constexpr int HW2  = HW * HW;    // 1048576

DI v8f wmma16(v16h a, v16h b, v8f c) {
  return __builtin_amdgcn_wmma_f32_16x16x32_f16(false, a, false, b, (short)0, c,
                                                false, false);
}

// A fragment (16x32 f16): lane's 16 halves are one contiguous run:
// K = half*16 + [0..15]  -> two 16B vector loads.
DI v16h load_a_row(const _Float16* __restrict__ R, int half) {
  h8 c0 = *(const h8*)(R + half * 16);
  h8 c1 = *(const h8*)(R + half * 16 + 8);
  v16h a;
#pragma unroll
  for (int t = 0; t < 8; ++t) { a[t] = c0[t]; a[t + 8] = c1[t]; }
  return a;
}

// B fragment (32x16): element t holds B[k0 + half*16 + t][n], column strided by ld.
DI v16h load_b_col_h(const _Float16* __restrict__ Bp, int ld, int half) {
  v16h b;
#pragma unroll
  for (int t = 0; t < 16; ++t) b[t] = Bp[(size_t)(half * 16 + t) * ld];
  return b;
}

DI float sigmoidf_(float x) { return 1.f / (1.f + __expf(-x)); }

// ---------------- small VALU kernels ----------------

__global__ void k_cvt_f16(const float* __restrict__ x, _Float16* __restrict__ y, int n) {
  int i = blockIdx.x * blockDim.x + threadIdx.x;
  if (i < n) y[i] = (_Float16)x[i];
}

// reorder 3x3 conv weights (o, c, kh, kw) -> (tap, o, c) f16
__global__ void k_cvt_w3(const float* __restrict__ w, _Float16* __restrict__ o, int n) {
  int i = blockIdx.x * blockDim.x + threadIdx.x;
  if (i >= n) return;
  int tap = i >> 12;
  int rem = i & 4095;
  int oc = rem >> 6;
  int c = rem & 63;
  o[i] = (_Float16)w[oc * 576 + c * 9 + tap];
}

// gap[b*C+c] = mean over HW
__global__ void k_gap(const float* __restrict__ feat, float* __restrict__ gap) {
  __shared__ float s[128];
  int row = blockIdx.x;
  const float* p = feat + (size_t)row * HW;
  float acc = 0.f;
  for (int i = threadIdx.x; i < HW; i += 128) acc += p[i];
  s[threadIdx.x] = acc;
  __syncthreads();
  for (int st = 64; st > 0; st >>= 1) {
    if (threadIdx.x < st) s[threadIdx.x] += s[threadIdx.x + st];
    __syncthreads();
  }
  if (threadIdx.x == 0) gap[row] = s[0] * (1.f / HW);
}

// lc[b,o] = sigmoid( sum_i lca_w[o,i,1]*gap[b,i] + lca_b[o] )  (len-1 conv, center tap only)
__global__ void k_lc(const float* __restrict__ gap, const float* __restrict__ w,
                     const float* __restrict__ bias, float* __restrict__ lc) {
  int i = blockIdx.x * blockDim.x + threadIdx.x;
  if (i >= Bn * Cc) return;
  int b = i / Cc, o = i % Cc;
  const float* g = gap + b * Cc;
  float acc = bias[o];
  for (int c = 0; c < Cc; ++c) acc += w[o * 768 + c * 3 + 1] * g[c];
  lc[i] = sigmoidf_(acc);
}

// gca q/k: 3-tap conv along channel axis of gap, then sigmoid
__global__ void k_gqk(const float* __restrict__ gap,
                      const float* __restrict__ qw, const float* __restrict__ qb,
                      const float* __restrict__ kw, const float* __restrict__ kb,
                      float* __restrict__ q, float* __restrict__ k) {
  int i = blockIdx.x * blockDim.x + threadIdx.x;
  if (i >= Bn * Cc) return;
  int b = i / Cc, c = i % Cc;
  const float* g = gap + b * Cc;
  float xm = (c > 0) ? g[c - 1] : 0.f;
  float x0 = g[c];
  float xp = (c < Cc - 1) ? g[c + 1] : 0.f;
  q[i] = sigmoidf_(qw[0] * xm + qw[1] * x0 + qw[2] * xp + qb[0]);
  k[i] = sigmoidf_(kw[0] * xm + kw[1] * x0 + kw[2] * xp + kb[0]);
}

// att_c[b,c,d] = softmax_c( k[b,c]*q[b,d] ), stored f16 (b,c,d), d fastest
__global__ void k_attc(const float* __restrict__ kg, const float* __restrict__ qg,
                       _Float16* __restrict__ attch) {
  __shared__ float s[256];
  int b = blockIdx.x >> 8;
  int d = blockIdx.x & 255;
  int c = threadIdx.x;
  float v = kg[b * Cc + c] * qg[b * Cc + d];
  s[c] = v;
  __syncthreads();
  for (int st = 128; st > 0; st >>= 1) {
    if (c < st) s[c] = fmaxf(s[c], s[c + st]);
    __syncthreads();
  }
  float m = s[0];
  __syncthreads();
  float e = __expf(v - m);
  s[c] = e;
  __syncthreads();
  for (int st = 128; st > 0; st >>= 1) {
    if (c < st) s[c] += s[c + st];
    __syncthreads();
  }
  attch[(size_t)b * CC + c * Cc + d] = (_Float16)(e / s[0]);
}

// row softmax over 1024 contiguous f32; result written only as f16 for the gs GEMM
__global__ void k_softmax_row(float* __restrict__ att, _Float16* __restrict__ atth) {
  __shared__ float s[256];
  size_t base = (size_t)blockIdx.x * HW;
  float* row = att + base;
  _Float16* rowh = atth + base;
  int t = threadIdx.x;
  float v[4];
  float m = -3.4e38f;
#pragma unroll
  for (int k = 0; k < 4; ++k) { v[k] = row[t + k * 256]; m = fmaxf(m, v[k]); }
  s[t] = m;
  __syncthreads();
  for (int st = 128; st > 0; st >>= 1) {
    if (t < st) s[t] = fmaxf(s[t], s[t + st]);
    __syncthreads();
  }
  m = s[0];
  __syncthreads();
  float sum = 0.f;
#pragma unroll
  for (int k = 0; k < 4; ++k) { v[k] = __expf(v[k] - m); sum += v[k]; }
  s[t] = sum;
  __syncthreads();
  for (int st = 128; st > 0; st >>= 1) {
    if (t < st) s[t] += s[t + st];
    __syncthreads();
  }
  float inv = 1.f / s[0];
#pragma unroll
  for (int k = 0; k < 4; ++k) rowh[t + k * 256] = (_Float16)(v[k] * inv);
}

// ls[b,p] = lsals_b + sum_{g,e} w[g*64+e]*f_g[b,e,p]
__global__ void k_ls(const _Float16* __restrict__ f1h, const _Float16* __restrict__ f3h,
                     const _Float16* __restrict__ f5h, const _Float16* __restrict__ f7h,
                     const float* __restrict__ w, const float* __restrict__ bias,
                     float* __restrict__ ls) {
  int i = blockIdx.x * blockDim.x + threadIdx.x;
  if (i >= Bn * HW) return;
  int b = i / HW, p = i % HW;
  const _Float16* f[4] = {f1h + (size_t)b * EHW, f3h + (size_t)b * EHW,
                          f5h + (size_t)b * EHW, f7h + (size_t)b * EHW};
  float acc = bias[0];
  for (int g = 0; g < 4; ++g)
    for (int e = 0; e < Ee; ++e)
      acc += w[g * Ee + e] * (float)f[g][e * HW + p];
  ls[i] = acc;
}

// elementwise fusion -> f16 for the final GEMM
__global__ void k_fused(const float* __restrict__ feat, const float* __restrict__ lc,
                        const float* __restrict__ ls, const float* __restrict__ gc,
                        const float* __restrict__ gsa, const float* __restrict__ fw,
                        _Float16* __restrict__ fusedh) {
  int i = blockIdx.x * blockDim.x + threadIdx.x;
  if (i >= Bn * CHW) return;
  int b = i / CHW;
  int r = i % CHW;
  int c = r / HW;
  int p = r % HW;
  float e0 = __expf(fw[0]), e1 = __expf(fw[1]), e2 = __expf(fw[2]);
  float inv = 1.f / (e0 + e1 + e2);
  float x = feat[i];
  float flc = x * lc[b * Cc + c] + x;
  float loc = flc * ls[b * HW + p] + flc;
  float fgc = x * gc[i];
  float glo = fgc * gsa[i] + fgc;
  fusedh[i] = (_Float16)((e0 * inv) * loc + (e1 * inv) * glo + (e2 * inv) * x);
}

// -------- WMMA GEMMs: one wave = 64x16 (MB=4) or 128x16 (MB=8) output tiles --------

// gc[b,n,d] = sum_c value[b,n,c]*att_c[b,c,d]; value = flat feature (b, n*256+c)
__global__ void k_gc(const _Float16* __restrict__ Fh, const _Float16* __restrict__ attch,
                     float* __restrict__ gc) {
  int lane = threadIdx.x, half = lane >> 4, l16 = lane & 15;
  int n0 = blockIdx.x * 16;   // d
  int m0 = blockIdx.y * 64;   // n
  int b = blockIdx.z;
  const _Float16* A = Fh + (size_t)b * CHW;     // A[n][c], contiguous in c
  const _Float16* Bm = attch + (size_t)b * CC;  // B[c][d], ld Cc
  v8f acc[4] = {};
  for (int k0 = 0; k0 < Cc; k0 += 32) {
    v16h bf = load_b_col_h(Bm + (size_t)k0 * Cc + n0 + l16, Cc, half);
#pragma unroll
    for (int mt = 0; mt < 4; ++mt) {
      v16h a = load_a_row(A + (m0 + mt * 16 + l16) * Cc + k0, half);
      acc[mt] = wmma16(a, bf, acc[mt]);
    }
  }
  float* O = gc + (size_t)b * CHW;
#pragma unroll
  for (int mt = 0; mt < 4; ++mt)
#pragma unroll
    for (int r = 0; r < 8; ++r)
      O[(m0 + mt * 16 + r + half * 8) * Cc + n0 + l16] = acc[mt][r];
}

// qs/ks/vs/f1: (64 x 256) @ (256 x HW) + bias -> f16; which==1 also writes ks^T (i,e)
__global__ void k_ecgemm(const _Float16* __restrict__ Fh,
                         const _Float16* wqh, const _Float16* wkh,
                         const _Float16* wvh, const _Float16* w1h,
                         const float* bq, const float* bk, const float* bv, const float* b1,
                         _Float16* oq, _Float16* ok, _Float16* ov, _Float16* o1,
                         _Float16* ksT) {
  int lane = threadIdx.x, half = lane >> 4, l16 = lane & 15;
  int n0 = blockIdx.x * 16;
  int which = blockIdx.z & 3, b = blockIdx.z >> 2;
  const _Float16* W = (which == 0) ? wqh : (which == 1) ? wkh : (which == 2) ? wvh : w1h;
  const float* bias = (which == 0) ? bq : (which == 1) ? bk : (which == 2) ? bv : b1;
  _Float16* out     = (which == 0) ? oq : (which == 1) ? ok : (which == 2) ? ov : o1;
  const _Float16* X = Fh + (size_t)b * CHW;   // B[c][i], ld HW
  v8f acc[4] = {};
  for (int k0 = 0; k0 < Cc; k0 += 32) {
    v16h bf = load_b_col_h(X + (size_t)k0 * HW + n0 + l16, HW, half);
#pragma unroll
    for (int mt = 0; mt < 4; ++mt) {
      v16h a = load_a_row(W + (mt * 16 + l16) * Cc + k0, half);
      acc[mt] = wmma16(a, bf, acc[mt]);
    }
  }
  _Float16* O = out + (size_t)b * EHW;
  _Float16* T = ksT + (size_t)b * EHW;
#pragma unroll
  for (int mt = 0; mt < 4; ++mt)
#pragma unroll
    for (int r = 0; r < 8; ++r) {
      int m = mt * 16 + r + half * 8;
      _Float16 v = (_Float16)(acc[mt][r] + bias[m]);
      O[m * HW + n0 + l16] = v;
      if (which == 1) T[(n0 + l16) * Ee + m] = v;
    }
}

// f3/f5/f7: dilated 3x3 conv on f1 as 9 taps of (64x64) GEMM; weights in (tap,o,c) f16
__global__ void k_conv3(const _Float16* __restrict__ f1h,
                        const _Float16* w3h, const _Float16* w5h, const _Float16* w7h,
                        const float* b3, const float* b5, const float* b7,
                        _Float16* o3, _Float16* o5, _Float16* o7) {
  int lane = threadIdx.x, half = lane >> 4, l16 = lane & 15;
  int n0 = blockIdx.x * 16;
  int which = blockIdx.z % 3, b = blockIdx.z / 3;
  int dil = which + 1;
  const _Float16* W = (which == 0) ? w3h : (which == 1) ? w5h : w7h;
  const float* bias = (which == 0) ? b3 : (which == 1) ? b5 : b7;
  _Float16* out     = (which == 0) ? o3 : (which == 1) ? o5 : o7;
  const _Float16* X = f1h + (size_t)b * EHW;
  int p = n0 + l16, py0 = p >> 5, px0 = p & 31;
  v8f acc[4] = {};
  for (int kh = 0; kh < 3; ++kh)
    for (int kw = 0; kw < 3; ++kw) {
      int py = py0 + dil * (kh - 1), px = px0 + dil * (kw - 1);
      bool valid = (py >= 0) && (py < Hh) && (px >= 0) && (px < Ww);
      int pp = valid ? (py * Ww + px) : 0;
      const _Float16* Wt = W + (kh * 3 + kw) * (Ee * Ee);
      for (int k0 = 0; k0 < Ee; k0 += 32) {
        v16h bf;
#pragma unroll
        for (int t = 0; t < 16; ++t) {
          _Float16 xv = X[(k0 + half * 16 + t) * HW + pp];
          bf[t] = valid ? xv : (_Float16)0.f;
        }
#pragma unroll
        for (int mt = 0; mt < 4; ++mt) {
          v16h a = load_a_row(Wt + (mt * 16 + l16) * Ee + k0, half);
          acc[mt] = wmma16(a, bf, acc[mt]);
        }
      }
    }
  _Float16* O = out + (size_t)b * EHW;
#pragma unroll
  for (int mt = 0; mt < 4; ++mt)
#pragma unroll
    for (int r = 0; r < 8; ++r) {
      int m = mt * 16 + r + half * 8;
      O[m * HW + n0 + l16] = (_Float16)(acc[mt][r] + bias[m]);
    }
}

// att_s logits: s[b,i,j] = sum_e ks[b,e,i]*qs[b,e,j]; A from ks^T (i-major, e contiguous)
__global__ void k_atts(const _Float16* __restrict__ ksT, const _Float16* __restrict__ qsh,
                       float* __restrict__ att) {
  int lane = threadIdx.x, half = lane >> 4, l16 = lane & 15;
  int n0 = blockIdx.x * 16;   // j
  int m0 = blockIdx.y * 64;   // i
  int b = blockIdx.z;
  const _Float16* A = ksT + (size_t)b * EHW;  // A[i][e], ld Ee
  const _Float16* Q = qsh + (size_t)b * EHW;  // B[e][j], ld HW
  v8f acc[4] = {};
  for (int k0 = 0; k0 < Ee; k0 += 32) {
    v16h bf = load_b_col_h(Q + (size_t)k0 * HW + n0 + l16, HW, half);
#pragma unroll
    for (int mt = 0; mt < 4; ++mt) {
      v16h a = load_a_row(A + (m0 + mt * 16 + l16) * Ee + k0, half);
      acc[mt] = wmma16(a, bf, acc[mt]);
    }
  }
  float* O = att + (size_t)b * HW2;
#pragma unroll
  for (int mt = 0; mt < 4; ++mt)
#pragma unroll
    for (int r = 0; r < 8; ++r)
      O[(size_t)(m0 + mt * 16 + r + half * 8) * HW + n0 + l16] = acc[mt][r];
}

// gs[b,e,j] = sum_i vs[b,e,i]*att_s[b,i,j]  (K=1024, whole E in one wave, f16 B)
__global__ void k_gs(const _Float16* __restrict__ vsh, const _Float16* __restrict__ atth,
                     _Float16* __restrict__ gsh) {
  int lane = threadIdx.x, half = lane >> 4, l16 = lane & 15;
  int n0 = blockIdx.x * 16;   // j
  int b = blockIdx.z;
  const _Float16* V = vsh + (size_t)b * EHW;   // A[e][i], contiguous in i
  const _Float16* A = atth + (size_t)b * HW2;  // B[i][j], ld HW
  v8f acc[4] = {};
  for (int k0 = 0; k0 < HW; k0 += 32) {
    if (k0 + 32 < HW)
      __builtin_prefetch(&A[(size_t)(k0 + 32 + half * 16) * HW + n0 + l16], 0, 0);
    v16h bf = load_b_col_h(A + (size_t)k0 * HW + n0 + l16, HW, half);
#pragma unroll
    for (int mt = 0; mt < 4; ++mt) {
      v16h a = load_a_row(V + (mt * 16 + l16) * HW + k0, half);
      acc[mt] = wmma16(a, bf, acc[mt]);
    }
  }
  _Float16* O = gsh + (size_t)b * EHW;
#pragma unroll
  for (int mt = 0; mt < 4; ++mt)
#pragma unroll
    for (int r = 0; r < 8; ++r) {
      int m = mt * 16 + r + half * 8;
      O[m * HW + n0 + l16] = (_Float16)acc[mt][r];
    }
}

// gsa_out[b,o,p] = gsa_ab[o] + sum_e gsa_aw[o,e]*gs[b,e,p]  (K=64, MB=8) -> f32
__global__ void k_gsaout(const _Float16* __restrict__ gsh, const _Float16* __restrict__ awh,
                         const float* __restrict__ bias, float* __restrict__ out) {
  int lane = threadIdx.x, half = lane >> 4, l16 = lane & 15;
  int n0 = blockIdx.x * 16;
  int m0 = blockIdx.y * 128;
  int b = blockIdx.z;
  const _Float16* X = gsh + (size_t)b * EHW;
  v8f acc[8] = {};
  for (int k0 = 0; k0 < Ee; k0 += 32) {
    v16h bf = load_b_col_h(X + (size_t)k0 * HW + n0 + l16, HW, half);
#pragma unroll
    for (int mt = 0; mt < 8; ++mt) {
      v16h a = load_a_row(awh + (m0 + mt * 16 + l16) * Ee + k0, half);
      acc[mt] = wmma16(a, bf, acc[mt]);
    }
  }
  float* O = out + (size_t)b * CHW;
#pragma unroll
  for (int mt = 0; mt < 8; ++mt)
#pragma unroll
    for (int r = 0; r < 8; ++r) {
      int m = m0 + mt * 16 + r + half * 8;
      O[m * HW + n0 + l16] = acc[mt][r] + bias[m];
    }
}

// final: out[b,o,p] = out_b[o] + sum_c out_w[o,c]*fused[b,c,p]  (K=256, MB=8) -> f32
__global__ void k_out(const _Float16* __restrict__ fusedh, const _Float16* __restrict__ owh,
                      const float* __restrict__ bias, float* __restrict__ out) {
  int lane = threadIdx.x, half = lane >> 4, l16 = lane & 15;
  int n0 = blockIdx.x * 16;
  int m0 = blockIdx.y * 128;
  int b = blockIdx.z;
  const _Float16* X = fusedh + (size_t)b * CHW;
  v8f acc[8] = {};
  for (int k0 = 0; k0 < Cc; k0 += 32) {
    v16h bf = load_b_col_h(X + (size_t)k0 * HW + n0 + l16, HW, half);
#pragma unroll
    for (int mt = 0; mt < 8; ++mt) {
      v16h a = load_a_row(owh + (m0 + mt * 16 + l16) * Cc + k0, half);
      acc[mt] = wmma16(a, bf, acc[mt]);
    }
  }
  float* O = out + (size_t)b * (CO * HW);
#pragma unroll
  for (int mt = 0; mt < 8; ++mt)
#pragma unroll
    for (int r = 0; r < 8; ++r) {
      int m = m0 + mt * 16 + r + half * 8;
      O[m * HW + n0 + l16] = acc[mt][r] + bias[m];
    }
}

// ---------------- launch ----------------

extern "C" void kernel_launch(void* const* d_in, const int* in_sizes, int n_in,
                              void* d_out, int out_size, void* d_ws, size_t ws_size,
                              hipStream_t stream) {
  (void)in_sizes; (void)n_in; (void)out_size; (void)ws_size;
  const float* feature = (const float*)d_in[0];
  const float* gca_qw = (const float*)d_in[1];
  const float* gca_qb = (const float*)d_in[2];
  const float* gca_kw = (const float*)d_in[3];
  const float* gca_kb = (const float*)d_in[4];
  const float* gsa_qw = (const float*)d_in[5];
  const float* gsa_qb = (const float*)d_in[6];
  const float* gsa_kw = (const float*)d_in[7];
  const float* gsa_kb = (const float*)d_in[8];
  const float* gsa_vw = (const float*)d_in[9];
  const float* gsa_vb = (const float*)d_in[10];
  const float* gsa_aw = (const float*)d_in[11];
  const float* gsa_ab = (const float*)d_in[12];
  const float* lca_w  = (const float*)d_in[13];
  const float* lca_b  = (const float*)d_in[14];
  const float* lsa1_w = (const float*)d_in[15];
  const float* lsa1_b = (const float*)d_in[16];
  const float* lsa3_w = (const float*)d_in[17];
  const float* lsa3_b = (const float*)d_in[18];
  const float* lsa5_w = (const float*)d_in[19];
  const float* lsa5_b = (const float*)d_in[20];
  const float* lsa7_w = (const float*)d_in[21];
  const float* lsa7_b = (const float*)d_in[22];
  const float* lsals_w = (const float*)d_in[23];
  const float* lsals_b = (const float*)d_in[24];
  const float* fusion_w = (const float*)d_in[25];
  const float* out_w  = (const float*)d_in[26];
  const float* out_b  = (const float*)d_in[27];
  float* out = (float*)d_out;

  // workspace carve-out
  char* base = (char*)d_ws;
  size_t off = 0;
  auto alloc = [&](size_t bytes) {
    size_t o = off;
    off = (off + bytes + 255) & ~(size_t)255;
    return (void*)(base + o);
  };
  float* gap   = (float*)alloc((size_t)Bn * Cc * 4);
  float* lc    = (float*)alloc((size_t)Bn * Cc * 4);
  float* qg    = (float*)alloc((size_t)Bn * Cc * 4);
  float* kg    = (float*)alloc((size_t)Bn * Cc * 4);
  float* ls    = (float*)alloc((size_t)Bn * HW * 4);
  _Float16* attch = (_Float16*)alloc((size_t)Bn * CC * 2);
  _Float16* Fh = (_Float16*)alloc((size_t)Bn * CHW * 2);
  float* gc    = (float*)alloc((size_t)Bn * CHW * 4);
  _Float16* qsh = (_Float16*)alloc((size_t)Bn * EHW * 2);
  _Float16* ksh = (_Float16*)alloc((size_t)Bn * EHW * 2);
  _Float16* vsh = (_Float16*)alloc((size_t)Bn * EHW * 2);
  _Float16* ksT = (_Float16*)alloc((size_t)Bn * EHW * 2);
  _Float16* f1h = (_Float16*)alloc((size_t)Bn * EHW * 2);
  _Float16* f3h = (_Float16*)alloc((size_t)Bn * EHW * 2);
  _Float16* f5h = (_Float16*)alloc((size_t)Bn * EHW * 2);
  _Float16* f7h = (_Float16*)alloc((size_t)Bn * EHW * 2);
  _Float16* gsh = (_Float16*)alloc((size_t)Bn * EHW * 2);
  float* atts     = (float*)alloc((size_t)Bn * HW2 * 4);
  _Float16* attsh = (_Float16*)alloc((size_t)Bn * HW2 * 2);
  float* gsaout = (float*)alloc((size_t)Bn * CHW * 4);
  _Float16* fusedh = (_Float16*)alloc((size_t)Bn * CHW * 2);
  // f16 weights
  _Float16* wqh = (_Float16*)alloc((size_t)Ee * Cc * 2);
  _Float16* wkh = (_Float16*)alloc((size_t)Ee * Cc * 2);
  _Float16* wvh = (_Float16*)alloc((size_t)Ee * Cc * 2);
  _Float16* w1h = (_Float16*)alloc((size_t)Ee * Cc * 2);
  _Float16* awh = (_Float16*)alloc((size_t)Cc * Ee * 2);
  _Float16* owh = (_Float16*)alloc((size_t)CO * Cc * 2);
  _Float16* w3h = (_Float16*)alloc((size_t)9 * Ee * Ee * 2);
  _Float16* w5h = (_Float16*)alloc((size_t)9 * Ee * Ee * 2);
  _Float16* w7h = (_Float16*)alloc((size_t)9 * Ee * Ee * 2);

  const int nFeat = Bn * CHW;
  const int nEC = Ee * Cc;        // 16384
  const int nW3 = 9 * Ee * Ee;    // 36864

  // stage 0: conversions and channel stats
  k_cvt_f16<<<dim3((nFeat + 255) / 256), dim3(256), 0, stream>>>(feature, Fh, nFeat);
  k_cvt_f16<<<dim3((nEC + 255) / 256), dim3(256), 0, stream>>>(gsa_qw, wqh, nEC);
  k_cvt_f16<<<dim3((nEC + 255) / 256), dim3(256), 0, stream>>>(gsa_kw, wkh, nEC);
  k_cvt_f16<<<dim3((nEC + 255) / 256), dim3(256), 0, stream>>>(gsa_vw, wvh, nEC);
  k_cvt_f16<<<dim3((nEC + 255) / 256), dim3(256), 0, stream>>>(lsa1_w, w1h, nEC);
  k_cvt_f16<<<dim3((nEC + 255) / 256), dim3(256), 0, stream>>>(gsa_aw, awh, nEC);
  k_cvt_f16<<<dim3((CO * Cc + 255) / 256), dim3(256), 0, stream>>>(out_w, owh, CO * Cc);
  k_cvt_w3<<<dim3((nW3 + 255) / 256), dim3(256), 0, stream>>>(lsa3_w, w3h, nW3);
  k_cvt_w3<<<dim3((nW3 + 255) / 256), dim3(256), 0, stream>>>(lsa5_w, w5h, nW3);
  k_cvt_w3<<<dim3((nW3 + 255) / 256), dim3(256), 0, stream>>>(lsa7_w, w7h, nW3);
  k_gap<<<dim3(Bn * Cc), dim3(128), 0, stream>>>(feature, gap);

  // stage 1: channel attentions (small) + gc einsum (WMMA)
  k_lc<<<dim3((Bn * Cc + 127) / 128), dim3(128), 0, stream>>>(gap, lca_w, lca_b, lc);
  k_gqk<<<dim3((Bn * Cc + 127) / 128), dim3(128), 0, stream>>>(gap, gca_qw, gca_qb,
                                                               gca_kw, gca_kb, qg, kg);
  k_attc<<<dim3(Bn * Cc), dim3(256), 0, stream>>>(kg, qg, attch);
  k_gc<<<dim3(Cc / 16, HW / 64, Bn), dim3(32), 0, stream>>>(Fh, attch, gc);

  // stage 2: 1x1 projections qs/ks/vs/f1 (WMMA)
  k_ecgemm<<<dim3(HW / 16, 1, Bn * 4), dim3(32), 0, stream>>>(
      Fh, wqh, wkh, wvh, w1h, gsa_qb, gsa_kb, gsa_vb, lsa1_b,
      qsh, ksh, vsh, f1h, ksT);

  // stage 3: dilated 3x3 convs (WMMA, 9 taps) + ls reduction
  k_conv3<<<dim3(HW / 16, 1, Bn * 3), dim3(32), 0, stream>>>(
      f1h, w3h, w5h, w7h, lsa3_b, lsa5_b, lsa7_b, f3h, f5h, f7h);
  k_ls<<<dim3((Bn * HW + 127) / 128), dim3(128), 0, stream>>>(
      f1h, f3h, f5h, f7h, lsals_w, lsals_b, ls);

  // stage 4: spatial attention (WMMA logits -> softmax -> WMMA apply -> WMMA proj)
  k_atts<<<dim3(HW / 16, HW / 64, Bn), dim3(32), 0, stream>>>(ksT, qsh, atts);
  k_softmax_row<<<dim3(Bn * HW), dim3(256), 0, stream>>>(atts, attsh);
  k_gs<<<dim3(HW / 16, 1, Bn), dim3(32), 0, stream>>>(vsh, attsh, gsh);
  k_gsaout<<<dim3(HW / 16, Cc / 128, Bn), dim3(32), 0, stream>>>(gsh, awh, gsa_ab, gsaout);

  // stage 5: elementwise fusion + final 1x1 conv (WMMA)
  k_fused<<<dim3((nFeat + 255) / 256), dim3(256), 0, stream>>>(
      feature, lc, ls, gc, gsaout, fusion_w, fusedh);
  k_out<<<dim3(HW / 16, CO / 128, Bn), dim3(32), 0, stream>>>(fusedh, owh, out_b, out);
}
